// VariationalGCNEncoder_89635967467597
// MI455X (gfx1250) — matrix operand order repaired and checked by reading, not compiled, not run on hardware
//
#include <hip/hip_runtime.h>

typedef _Float16 f16;
typedef __attribute__((ext_vector_type(16))) _Float16 v16h;
typedef __attribute__((ext_vector_type(8)))  _Float16 v8h;
typedef __attribute__((ext_vector_type(8)))  float    v8f;

#define TB 256   // 8 waves (wave32)

// ---------------- degree / normalization ----------------

__global__ void k_init_deg(float* __restrict__ deg, int n) {
  int i = blockIdx.x * blockDim.x + threadIdx.x;
  if (i < n) deg[i] = 1.0f;               // self-loop contribution
}

__global__ void k_deg_atomic(const int* __restrict__ dst, float* __restrict__ deg, int E) {
  int e = blockIdx.x * blockDim.x + threadIdx.x;
  if (e < E) unsafeAtomicAdd(&deg[dst[e]], 1.0f);
}

__global__ void k_dinv(float* __restrict__ deg, int n) {
  int i = blockIdx.x * blockDim.x + threadIdx.x;
  if (i < n) deg[i] = rsqrtf(deg[i]);     // deg >= 1 always (self-loops)
}

__global__ void k_norm(const int* __restrict__ src, const int* __restrict__ dst,
                       const float* __restrict__ dinv, float* __restrict__ norm, int E) {
  int e = blockIdx.x * blockDim.x + threadIdx.x;
  if (e < E) norm[e] = dinv[src[e]] * dinv[dst[e]];
}

// ---------------- conversions / packing ----------------

__global__ void k_f32_to_f16(const float* __restrict__ in, f16* __restrict__ out, int n) {
  int i = blockIdx.x * blockDim.x + threadIdx.x;
  if (i < n) out[i] = (f16)in[i];
}

__global__ void k_zero(float* __restrict__ p, int n) {
  int i = blockIdx.x * blockDim.x + threadIdx.x;
  if (i < n) p[i] = 0.0f;
}

// Pack a 128x128 (row-major, [K][N]) weight into WMMA B-fragment order:
// flat = ((ktile*8 + ntile)*32 + lane)*16 + h
//   n = ntile*16 + (lane&15)
//   k = ktile*32 + (lane>=16 ? 8 : 0) + (h&7) + (h>=8 ? 16 : 0)
// mode 0: Wa is 128x128.  mode 1: columns 0..63 from Wa (128x64), 64..127 from Wb.
__global__ void k_pack_w(const float* __restrict__ Wa, const float* __restrict__ Wb,
                         f16* __restrict__ out, int mode) {
  int tid = blockIdx.x * blockDim.x + threadIdx.x;
  if (tid >= 128 * 128) return;
  int h     = tid & 15;
  int lane  = (tid >> 4) & 31;
  int ntile = (tid >> 9) & 7;
  int ktile = (tid >> 12) & 3;
  int n = (ntile << 4) + (lane & 15);
  int k = (ktile << 5) + ((lane >> 4) << 3) + (h & 7) + ((h >> 3) << 4);
  float v;
  if (mode == 0)      v = Wa[k * 128 + n];
  else if (n < 64)    v = Wa[k * 64 + n];
  else                v = Wb[k * 64 + (n - 64)];
  out[tid] = (f16)v;
}

// ---------------- WMMA GEMM: C[M,128] = A[M,128] * B[128,128] ----------------
// A: row-major f16; Bf: fragment-packed f16; C: row-major f32.
// One block = 8 waves, each wave computes one 16x16 tile; block covers 16 rows x 128 cols.
__global__ void k_gemm_wmma(const f16* __restrict__ A, const f16* __restrict__ Bf,
                            float* __restrict__ C, int M) {
  const int lane  = threadIdx.x & 31;
  const int ntile = threadIdx.x >> 5;        // 0..7
  const int mtile = blockIdx.x;
  int rowa = (mtile << 4) + (lane & 15);     // A row this lane supplies
  if (rowa >= M) rowa = M - 1;               // keep EXEC all-ones for WMMA
  const int khalf = (lane >> 4) << 3;        // lanes 16..31 carry K+8 group
  const f16* arow = A + (size_t)rowa * 128 + khalf;

  v8f c = {};
#pragma unroll
  for (int kt = 0; kt < 4; ++kt) {
    // A fragment (16-bit 16x32): halves 0..7 = K kbase..kbase+7, halves 8..15 = K kbase+16..+23
    v8h a0 = *(const v8h*)(arow + kt * 32);
    v8h a1 = *(const v8h*)(arow + kt * 32 + 16);
    v16h a;
#pragma unroll
    for (int i = 0; i < 8; ++i) { a[i] = a0[i]; a[8 + i] = a1[i]; }
    v16h b = *(const v16h*)(Bf + (((kt << 3) + ntile) * 32 + lane) * 16);
    c = __builtin_amdgcn_wmma_f32_16x16x32_f16(false, a, false, b, (short)0, c,
                                               false, false);
  }

  // C layout: VGPR r -> M = r (lanes 0-15) / 8+r (lanes 16-31); N = lane&15
  const int mbase = (mtile << 4) + ((lane >> 4) << 3);
  const int n = (ntile << 4) + (lane & 15);
  float* crow = C + (size_t)mbase * 128 + n;
  if (((mtile << 4) + 16) <= M) {
    // uniform (scalar-branch) fast path: whole tile in range, no per-lane guards
#pragma unroll
    for (int r = 0; r < 8; ++r) crow[(size_t)r * 128] = c[r];
  } else {
#pragma unroll
    for (int r = 0; r < 8; ++r) {
      if (mbase + r < M) crow[(size_t)r * 128] = c[r];
    }
  }
}

// ---------------- edge scatter-add: agg[dst] += T[src] * norm ----------------
// one thread per (edge, 4 features)
__global__ void k_scatter(const float* __restrict__ T, const int* __restrict__ src,
                          const int* __restrict__ dst, const float* __restrict__ norm,
                          float* __restrict__ agg, int E) {
  int gid = blockIdx.x * blockDim.x + threadIdx.x;
  int e = gid >> 5;
  if (e >= E) return;
  int q = (gid & 31) << 2;
  int s = src[e], d = dst[e];
  float w = norm[e];
  const float4 v = *(const float4*)(T + (size_t)s * 128 + q);
  float* o = agg + (size_t)d * 128 + q;
  unsafeAtomicAdd(o + 0, v.x * w);
  unsafeAtomicAdd(o + 1, v.y * w);
  unsafeAtomicAdd(o + 2, v.z * w);
  unsafeAtomicAdd(o + 3, v.w * w);
}

// ---------------- finalize passes ----------------

// h = relu(agg + T*dinv^2 + b1) -> f16
__global__ void k_finalize1(const float* __restrict__ T, const float* __restrict__ agg,
                            const float* __restrict__ dinv, const float* __restrict__ b1,
                            f16* __restrict__ Hh, int n) {
  int i = blockIdx.x * blockDim.x + threadIdx.x;
  if (i >= n) return;
  int node = i >> 7, c = i & 127;
  float di = dinv[node];
  float v = agg[i] + T[i] * di * di + b1[c];
  Hh[i] = (f16)fmaxf(v, 0.0f);
}

// mu / logstd written into d_out: [0,N*64) = mu, [N*64, 2*N*64) = logstd
__global__ void k_finalize2(const float* __restrict__ T, const float* __restrict__ agg,
                            const float* __restrict__ dinv, const float* __restrict__ bmu,
                            const float* __restrict__ bls, float* __restrict__ out,
                            int N, int n) {
  int i = blockIdx.x * blockDim.x + threadIdx.x;
  if (i >= n) return;
  int node = i >> 7, c = i & 127;
  float di = dinv[node];
  float v = agg[i] + T[i] * di * di;
  if (c < 64) out[(size_t)node * 64 + c] = v + bmu[c];
  else        out[(size_t)N * 64 + (size_t)node * 64 + (c - 64)] = v + bls[c - 64];
}

// ---------------- host launcher ----------------

extern "C" void kernel_launch(void* const* d_in, const int* in_sizes, int n_in,
                              void* d_out, int out_size, void* d_ws, size_t ws_size,
                              hipStream_t stream) {
  const float* x   = (const float*)d_in[0];
  const int*   ei  = (const int*)d_in[1];
  const float* W1  = (const float*)d_in[2];
  const float* b1  = (const float*)d_in[3];
  const float* Wmu = (const float*)d_in[4];
  const float* bmu = (const float*)d_in[5];
  const float* Wls = (const float*)d_in[6];
  const float* bls = (const float*)d_in[7];

  const int N = in_sizes[0] / 128;
  const int E = in_sizes[1] / 2;
  const int* src = ei;
  const int* dst = ei + E;

  auto al = [](size_t v) { return (v + 255) & ~(size_t)255; };
  char* p = (char*)d_ws;
  float* dinv = (float*)p;  p += al((size_t)N * 4);
  float* norm = (float*)p;  p += al((size_t)E * 4);
  f16* w1f    = (f16*)p;    p += al((size_t)128 * 128 * 2);
  f16* wcf    = (f16*)p;    p += al((size_t)128 * 128 * 2);
  f16* xh     = (f16*)p;    p += al((size_t)N * 128 * 2);   // reused as Hh after layer 1
  float* T    = (float*)p;  p += al((size_t)N * 128 * 4);   // GEMM output (both layers)
  float* agg  = (float*)p;  p += al((size_t)N * 128 * 4);   // scatter accumulator

  const int NF = N * 128;
  const int gN   = (N + TB - 1) / TB;
  const int gE   = (E + TB - 1) / TB;
  const int gNF  = (NF + TB - 1) / TB;
  const int gSc  = (E * 32 + TB - 1) / TB;
  const int gMt  = (N + 15) / 16;
  const int gPk  = (128 * 128 + TB - 1) / TB;

  // normalization
  k_init_deg<<<gN, TB, 0, stream>>>(dinv, N);
  k_deg_atomic<<<gE, TB, 0, stream>>>(dst, dinv, E);
  k_dinv<<<gN, TB, 0, stream>>>(dinv, N);
  k_norm<<<gE, TB, 0, stream>>>(src, dst, dinv, norm, E);

  // precision conversion + weight packing
  k_f32_to_f16<<<gNF, TB, 0, stream>>>(x, xh, NF);
  k_pack_w<<<gPk, TB, 0, stream>>>(W1, W1, w1f, 0);
  k_pack_w<<<gPk, TB, 0, stream>>>(Wmu, Wls, wcf, 1);

  // ---- layer 1: T = x@W1 ; agg = scatter ; h = relu(agg + self + b1) ----
  k_zero<<<gNF, TB, 0, stream>>>(agg, NF);
  k_gemm_wmma<<<gMt, TB, 0, stream>>>(xh, w1f, T, N);
  k_scatter<<<gSc, TB, 0, stream>>>(T, src, dst, norm, agg, E);
  k_finalize1<<<gNF, TB, 0, stream>>>(T, agg, dinv, b1, xh /* Hh alias */, NF);

  // ---- layer 2+3 fused: T = h@[Wmu|Wls] ; agg = scatter ; out = agg + self + bias ----
  k_zero<<<gNF, TB, 0, stream>>>(agg, NF);
  k_gemm_wmma<<<gMt, TB, 0, stream>>>(xh, wcf, T, N);
  k_scatter<<<gSc, TB, 0, stream>>>(T, src, dst, norm, agg, E);
  k_finalize2<<<gNF, TB, 0, stream>>>(T, agg, dinv, bmu, bls, (float*)d_out, N, NF);

  (void)n_in; (void)out_size; (void)ws_size;
}
